// hDRMLPv3Embed_70076686402002
// MI455X (gfx1250) — matrix operand order repaired and checked by reading, not compiled, not run on hardware
//
#include <hip/hip_runtime.h>
#include <math.h>

#define DEVI __device__ __forceinline__

typedef __bf16 v16bf __attribute__((ext_vector_type(16)));
typedef float  v8f   __attribute__((ext_vector_type(8)));

union ABFrag { v16bf v; unsigned int u[8]; };

#if __has_builtin(__builtin_amdgcn_global_load_async_to_lds_b128) && __has_builtin(__builtin_amdgcn_s_wait_asynccnt)
#define USE_ASYNC_LDS 1
typedef int v4i_t __attribute__((vector_size(16)));
typedef __attribute__((address_space(1))) v4i_t* gptr_v4i;
typedef __attribute__((address_space(3))) v4i_t* lptr_v4i;
#else
#define USE_ASYNC_LDS 0
#endif

DEVI float gelu_f(float x) { return 0.5f * x * (1.0f + erff(x * 0.70710678118654752f)); }

DEVI unsigned short f2bf(float f) {
    union { float f; unsigned int u; } c; c.f = f;
    unsigned int u = c.u + 0x7FFFu + ((c.u >> 16) & 1u);   // round-to-nearest-even
    return (unsigned short)(u >> 16);
}

DEVI float bn_apply(float x, float s, float b, float m, float v) {
    float sc = s * rsqrtf(v + 1e-5f);
    return x * sc + (b - m * sc);
}

// ---------------------------------------------------------------------------
// Kernel 1: fc conv 3x3, 3->48, SAME padding, + bias.  y: [16,48,224,224] f32
// ---------------------------------------------------------------------------
__global__ void k_conv_fc(const float* __restrict__ x, const float* __restrict__ w,
                          const float* __restrict__ bias, float* __restrict__ y) {
    int idx = blockIdx.x * blockDim.x + threadIdx.x;
    const int TOT = 16 * 48 * 224 * 224;
    if (idx >= TOT) return;
    int gx = idx % 224; int t = idx / 224;
    int gy = t % 224;   t /= 224;
    int c  = t % 48;    int b = t / 48;
    float acc = bias[c];
    for (int ci = 0; ci < 3; ++ci)
        for (int ky = 0; ky < 3; ++ky) {
            int yy = gy + ky - 1; if (yy < 0 || yy >= 224) continue;
            for (int kx = 0; kx < 3; ++kx) {
                int xx = gx + kx - 1; if (xx < 0 || xx >= 224) continue;
                acc += x[((size_t)(b * 3 + ci) * 224 + yy) * 224 + xx] *
                       w[((c * 3 + ci) * 3 + ky) * 3 + kx];
            }
        }
    y[idx] = acc;
}

// ---------------------------------------------------------------------------
// Branch stage A: a = gelu(bn1(x)); also write bf16 x into stacked-K buffer
// S[b][r][pix][C + c], zero K-padding region.
// ---------------------------------------------------------------------------
__global__ void k_branch_pre(const float* __restrict__ in,
                             const float* __restrict__ s1, const float* __restrict__ b1,
                             const float* __restrict__ m1, const float* __restrict__ v1,
                             float* __restrict__ abuf, unsigned short* __restrict__ S,
                             int C, int gw, int tile, int Kpad) {
    int idx = blockIdx.x * blockDim.x + threadIdx.x;
    int TOT = 16 * C * 224 * 224;
    if (idx >= TOT) return;
    int gx = idx % 224; int t = idx / 224;
    int gy = t % 224;   t /= 224;
    int c  = t % C;     int b = t / C;
    int ry = gy / tile, rx = gx / tile;
    int r  = ry * gw + rx;
    int pc = r * C + c;
    float xv = in[idx];
    float av = gelu_f(bn_apply(xv, s1[pc], b1[pc], m1[pc], v1[pc]));
    abuf[idx] = av;
    int p    = (gy - ry * tile) * tile + (gx - rx * tile);
    int R    = gw * gw;
    int npix = tile * tile;
    size_t base = ((size_t)(b * R + r) * npix + p) * Kpad;
    S[base + C + c] = f2bf(xv);
    int padn = Kpad - 2 * C;
    if (c < padn) S[base + 2 * C + c] = 0;
}

// ---------------------------------------------------------------------------
// Branch stage B: depthwise 3x3 (zero-padded at region tile edges),
// t = gelu(bn2(.)) -> S[b][r][pix][c] (bf16)
// ---------------------------------------------------------------------------
__global__ void k_branch_dw(const float* __restrict__ abuf, const float* __restrict__ dww,
                            const float* __restrict__ s2, const float* __restrict__ b2,
                            const float* __restrict__ m2, const float* __restrict__ v2,
                            unsigned short* __restrict__ S,
                            int C, int gw, int tile, int Kpad) {
    int idx = blockIdx.x * blockDim.x + threadIdx.x;
    int TOT = 16 * C * 224 * 224;
    if (idx >= TOT) return;
    int gx = idx % 224; int t = idx / 224;
    int gy = t % 224;   t /= 224;
    int c  = t % C;     int b = t / C;
    int ry = gy / tile, rx = gx / tile;
    int r  = ry * gw + rx;
    int pc = r * C + c;
    int y0 = ry * tile, x0 = rx * tile;
    const float* wp = dww + (size_t)pc * 9;
    float acc = 0.f;
    for (int dy = -1; dy <= 1; ++dy) {
        int yy = gy + dy; if (yy < y0 || yy >= y0 + tile) continue;
        for (int dx = -1; dx <= 1; ++dx) {
            int xx = gx + dx; if (xx < x0 || xx >= x0 + tile) continue;
            acc += wp[(dy + 1) * 3 + (dx + 1)] *
                   abuf[((size_t)(b * C + c) * 224 + yy) * 224 + xx];
        }
    }
    float tv = gelu_f(bn_apply(acc, s2[pc], b2[pc], m2[pc], v2[pc]));
    int p = (gy - y0) * tile + (gx - x0);
    S[((size_t)(b * gw * gw + r) * (tile * tile) + p) * Kpad + c] = f2bf(tv);
}

// ---------------------------------------------------------------------------
// Pre-pack branch weights into WMMA B-fragment layout.
// W[k][o]: k<C -> pw_w[r][o][k]; C<=k<2C -> res_w[r][o][k-C]; else 0.
// ---------------------------------------------------------------------------
__global__ void k_prepack_branch(const float* __restrict__ pw, const float* __restrict__ rw,
                                 unsigned int* __restrict__ Wf,
                                 int R, int C, int O, int Kpad, int Opad) {
    int KT = Kpad / 32, NT = Opad / 16;
    int tot = R * KT * NT * 256;
    int idx = blockIdx.x * blockDim.x + threadIdx.x;
    if (idx >= tot) return;
    int p = idx & 7; int lane = (idx >> 3) & 31; int t = idx >> 8;
    int nt = t % NT; t /= NT; int kt = t % KT; int r = t / KT;
    int o  = nt * 16 + (lane & 15);
    int kb = kt * 32 + ((lane < 16) ? 0 : 8) + ((p < 4) ? 2 * p : 16 + 2 * (p - 4));
    float w0 = 0.f, w1 = 0.f;
    if (o < O) {
        int k0 = kb, k1 = kb + 1;
        w0 = (k0 < C) ? pw[((size_t)r * O + o) * C + k0]
           : (k0 < 2 * C) ? rw[((size_t)r * O + o) * C + (k0 - C)] : 0.f;
        w1 = (k1 < C) ? pw[((size_t)r * O + o) * C + k1]
           : (k1 < 2 * C) ? rw[((size_t)r * O + o) * C + (k1 - C)] : 0.f;
    }
    Wf[idx] = (unsigned int)f2bf(w0) | ((unsigned int)f2bf(w1) << 16);
}

// ---------------------------------------------------------------------------
// Pre-pack patch-conv weight (OIHW) as GEMM B fragments, k = c*KH*KW + ky*KW + kx
// ---------------------------------------------------------------------------
__global__ void k_prepack_conv(const float* __restrict__ w, unsigned int* __restrict__ Wf,
                               int Cin, int KH, int KW, int O) {
    int K = Cin * KH * KW;
    int KT = K / 32, NT = O / 16;
    int tot = KT * NT * 256;
    int idx = blockIdx.x * blockDim.x + threadIdx.x;
    if (idx >= tot) return;
    int p = idx & 7; int lane = (idx >> 3) & 31; int t = idx >> 8;
    int nt = t % NT; int kt = t / NT;
    int o  = nt * 16 + (lane & 15);
    int kb = kt * 32 + ((lane < 16) ? 0 : 8) + ((p < 4) ? 2 * p : 16 + 2 * (p - 4));
    float vals[2];
    for (int j = 0; j < 2; ++j) {
        int k  = kb + j;
        int kx = k % KW; int tt = k / KW; int ky = tt % KH; int c = tt / KH;
        vals[j] = w[(((size_t)o * Cin + c) * KH + ky) * KW + kx];
    }
    Wf[idx] = (unsigned int)f2bf(vals[0]) | ((unsigned int)f2bf(vals[1]) << 16);
}

// ---------------------------------------------------------------------------
// Branch GEMM via v_wmma_f32_16x16x32_bf16 (per (b,r) tiny-K GEMMs).
// ---------------------------------------------------------------------------
__global__ void __launch_bounds__(256) k_gemm_branch(
        const unsigned short* __restrict__ S, const unsigned int* __restrict__ Wf,
        const float* __restrict__ pwb, const float* __restrict__ resb,
        float* __restrict__ out,
        int R, int C, int O, int Kpad, int Opad, int gw, int tile) {
    int npix = tile * tile;
    int MT = npix / 16, NT = Opad / 16, KT = Kpad / 32;
    int wavesPerB = R * MT * NT;
    int tid  = blockIdx.x * 8 + (threadIdx.x >> 5);
    int lane = threadIdx.x & 31;
    if (tid >= 16 * wavesPerB) return;       // wave-uniform exit (EXEC stays full)
    int b = tid / wavesPerB; int t = tid % wavesPerB;
    int nt = t % NT; t /= NT; int mt = t % MT; int r = t / MT;

    int arow = mt * 16 + (lane & 15);
    const unsigned short* Ab = S + ((size_t)(b * R + r) * npix + arow) * Kpad;
    int klo = (lane < 16) ? 0 : 8;

    v8f acc = {0.f, 0.f, 0.f, 0.f, 0.f, 0.f, 0.f, 0.f};
    for (int kt = 0; kt < KT; ++kt) {
        ABFrag A, Bf;
        int kb = kt * 32 + klo;
#pragma unroll
        for (int j = 0; j < 4; ++j) {
            A.u[j]     = *(const unsigned int*)(Ab + kb + 2 * j);
            A.u[j + 4] = *(const unsigned int*)(Ab + kb + 16 + 2 * j);
        }
        const unsigned int* wsrc = Wf + (((size_t)(r * KT + kt) * NT + nt) * 32 + lane) * 8;
#pragma unroll
        for (int j = 0; j < 8; ++j) Bf.u[j] = wsrc[j];
        acc = __builtin_amdgcn_wmma_f32_16x16x32_bf16(false, A.v, false, Bf.v,
                                                      (short)0, acc, false, false);
    }
    int o = nt * 16 + (lane & 15);
    if (o < O) {
        int ry = r / gw, rx = r % gw;
        float bias = pwb[r * O + o] + resb[r * O + o];
        int mofs = (lane >= 16) ? 8 : 0;
#pragma unroll
        for (int i = 0; i < 8; ++i) {
            int m  = mt * 16 + i + mofs;
            int ly = m / tile, lx = m % tile;
            int gy = ry * tile + ly, gx = rx * tile + lx;
            out[(((size_t)b * O + o) * 224 + gy) * 224 + gx] = acc[i] + bias;
        }
    }
}

// ---------------------------------------------------------------------------
// Combine: z = gelu(bn_n1(y + concat(b1,b2,b3))); write bf16 straight into the
// conv2 GEMM A layout (k4s4 patches non-overlapping => im2col-free).
// ---------------------------------------------------------------------------
__global__ void k_combine(const float* __restrict__ y, const float* __restrict__ b1,
                          const float* __restrict__ b2, const float* __restrict__ b3,
                          const float* __restrict__ ns, const float* __restrict__ nb,
                          const float* __restrict__ nm, const float* __restrict__ nv,
                          unsigned short* __restrict__ A2) {
    int idx = blockIdx.x * blockDim.x + threadIdx.x;
    const int TOT = 16 * 48 * 224 * 224;
    if (idx >= TOT) return;
    int gx = idx % 224; int t = idx / 224;
    int gy = t % 224;   t /= 224;
    int c  = t % 48;    int b = t / 48;
    float add;
    if (c < 24)      add = b1[((size_t)(b * 24 + c) * 224 + gy) * 224 + gx];
    else if (c < 36) add = b2[((size_t)(b * 12 + (c - 24)) * 224 + gy) * 224 + gx];
    else             add = b3[((size_t)(b * 12 + (c - 36)) * 224 + gy) * 224 + gx];
    float z = gelu_f(bn_apply(y[idx] + add, ns[c], nb[c], nm[c], nv[c]));
    int pix = (gy >> 2) * 56 + (gx >> 2);
    int k   = c * 16 + (gy & 3) * 4 + (gx & 3);
    A2[((size_t)b * 3136 + pix) * 768 + k] = f2bf(z);
}

// ---------------------------------------------------------------------------
// Tail GEMM (K=768), LDS-staged weight fragments + 4 N-tiles per wave.
// Block = 8 waves sharing (b, n-group); wave w owns M-tile mtg*8+w.
// K processed in chunks of 8 WMMA steps; chunk weights staged to LDS (32 KB)
// via CDNA5 GLOBAL_LOAD_ASYNC_TO_LDS_B128 (fallback: load + ds_store).
// B fragments read back with ds_load_b128.
// mode 0: gelu(bn) -> scatter bf16 to next conv's A; mode 1: bn -> f32 d_out.
// ---------------------------------------------------------------------------
#define NTPW 4
#define KCHUNK 8
__global__ void __launch_bounds__(256) k_gemm_tail2(
        const unsigned short* __restrict__ A, const unsigned int* __restrict__ Wf,
        const float* __restrict__ s, const float* __restrict__ bb,
        const float* __restrict__ m_, const float* __restrict__ v_,
        unsigned short* __restrict__ nextA, float* __restrict__ fout,
        int Mrows, int MT, int O, int mode, int outW, int nextW, int nextRows,
        int Mvalid, int MTG) {
    __shared__ unsigned int Wlds[KCHUNK * NTPW * 256];   // 32 KB
    const int KT = 24;
    int NT  = O / 16;
    int NTG = NT / NTPW;
    int bid = blockIdx.x;
    int mtg = bid % MTG; int t = bid / MTG;
    int ntg = t % NTG;   int b = t / NTG;
    int wid = threadIdx.x >> 5, lane = threadIdx.x & 31;
    int mt = mtg * 8 + wid;
    bool active = (mt < MT);
    int arow = active ? (mt * 16 + (lane & 15)) : 0;
    const unsigned short* Ab = A + ((size_t)b * Mrows + arow) * 768;
    int klo = (lane < 16) ? 0 : 8;
    int nt0 = ntg * NTPW;

    v8f acc[NTPW];
#pragma unroll
    for (int j = 0; j < NTPW; ++j) acc[j] = (v8f){0.f,0.f,0.f,0.f,0.f,0.f,0.f,0.f};

    for (int kc = 0; kc < KT; kc += KCHUNK) {
        // stage KCHUNK x NTPW weight fragments (8192 dwords) into LDS
#pragma unroll
        for (int i = 0; i < 32; i += 4) {
            int f    = threadIdx.x * 32 + i;
            int elem = f & 255;
            int j    = (f >> 8) & (NTPW - 1);
            int kk   = f >> 10;
            size_t g = ((size_t)(kc + kk) * NT + nt0 + j) * 256 + elem;
            unsigned int* l = &Wlds[(kk * NTPW + j) * 256 + elem];
#if USE_ASYNC_LDS
            __builtin_amdgcn_global_load_async_to_lds_b128(
                (gptr_v4i)&Wf[g], (lptr_v4i)l, 0, 0);
#else
            *(uint4*)l = *(const uint4*)&Wf[g];
#endif
        }
#if USE_ASYNC_LDS
        __builtin_amdgcn_s_wait_asynccnt(0);
#endif
        __syncthreads();

        for (int kk = 0; kk < KCHUNK; ++kk) {
            ABFrag Af;
            int kb = (kc + kk) * 32 + klo;
#pragma unroll
            for (int q = 0; q < 4; ++q) {
                Af.u[q]     = *(const unsigned int*)(Ab + kb + 2 * q);
                Af.u[q + 4] = *(const unsigned int*)(Ab + kb + 16 + 2 * q);
            }
#pragma unroll
            for (int j = 0; j < NTPW; ++j) {
                ABFrag Bf;
                const unsigned int* wl = &Wlds[(kk * NTPW + j) * 256 + lane * 8];
#pragma unroll
                for (int q = 0; q < 8; ++q) Bf.u[q] = wl[q];
                acc[j] = __builtin_amdgcn_wmma_f32_16x16x32_bf16(
                             false, Af.v, false, Bf.v, (short)0, acc[j], false, false);
            }
        }
        __syncthreads();
    }

    if (active) {
        int mofs = (lane >= 16) ? 8 : 0;
#pragma unroll
        for (int j = 0; j < NTPW; ++j) {
            int o = (nt0 + j) * 16 + (lane & 15);
            float sc = s[o] * rsqrtf(v_[o] + 1e-5f);
            float sh = bb[o] - m_[o] * sc;
#pragma unroll
            for (int i = 0; i < 8; ++i) {
                int mm = mt * 16 + i + mofs;
                if (mm >= Mvalid) continue;
                float val = acc[j][i] * sc + sh;
                if (mode == 0) {
                    val = gelu_f(val);
                    int py = mm / outW, px = mm % outW;
                    int pix3 = (py >> 1) * nextW + (px >> 1);
                    int k3   = o * 4 + (py & 1) * 2 + (px & 1);
                    nextA[((size_t)b * nextRows + pix3) * 768 + k3] = f2bf(val);
                } else {
                    fout[((size_t)b * Mvalid + mm) * 768 + o] = val;
                }
            }
        }
    }
}

// ---------------------------------------------------------------------------
extern "C" void kernel_launch(void* const* d_in, const int* in_sizes, int n_in,
                              void* d_out, int out_size, void* d_ws, size_t ws_size,
                              hipStream_t stream) {
    (void)in_sizes; (void)n_in; (void)out_size; (void)ws_size;
    const float* x   = (const float*)d_in[0];
    const float* fcw = (const float*)d_in[1];
    const float* fcb = (const float*)d_in[2];
    const int B1 = 3, B2 = 16, B3 = 29;
    auto br = [&](int base, int k) { return (const float*)d_in[base + k]; };
    const float* n1b = (const float*)d_in[42]; const float* n1m = (const float*)d_in[43];
    const float* n1s = (const float*)d_in[44]; const float* n1v = (const float*)d_in[45];
    const float* c2w = (const float*)d_in[46];
    const float* bn2b = (const float*)d_in[47]; const float* bn2m = (const float*)d_in[48];
    const float* bn2s = (const float*)d_in[49]; const float* bn2v = (const float*)d_in[50];
    const float* c3w = (const float*)d_in[51];
    const float* bn3b = (const float*)d_in[52]; const float* bn3m = (const float*)d_in[53];
    const float* bn3s = (const float*)d_in[54]; const float* bn3v = (const float*)d_in[55];
    const float* c4w = (const float*)d_in[56];
    const float* bn4b = (const float*)d_in[57]; const float* bn4m = (const float*)d_in[58];
    const float* bn4s = (const float*)d_in[59]; const float* bn4v = (const float*)d_in[60];

    // ---- workspace partition (deterministic) ----
    char* ws = (char*)d_ws;
    auto al = [](size_t x) { return (x + 255) & ~(size_t)255; };
    const size_t SZ_Y  = (size_t)16 * 48 * 224 * 224 * 4;
    const size_t SZ_S  = (size_t)16 * 64 * 784 * 96 * 2;
    const size_t SZ_B1 = (size_t)16 * 24 * 224 * 224 * 4;
    const size_t SZ_B2 = (size_t)16 * 12 * 224 * 224 * 4;
    size_t off = 0;
    size_t off_y  = off; off += al(SZ_Y);
    size_t off_a  = off; off += al(SZ_Y);
    size_t off_S  = off; off += al(SZ_S);
    size_t off_b1 = off; off += al(SZ_B1);
    size_t off_b2 = off; off += al(SZ_B2);
    size_t off_b3 = off; off += al(SZ_B2);
    size_t off_wf1 = off; off += al((size_t)64 * 3 * 2 * 256 * 4);
    size_t off_wf2 = off; off += al((size_t)16 * 2 * 1 * 256 * 4);
    size_t off_wf3 = off; off += al((size_t)4 * 1 * 1 * 256 * 4);
    size_t off_wc2 = off; off += al((size_t)24 * 12 * 256 * 4);
    size_t off_wc3 = off; off += al((size_t)24 * 12 * 256 * 4);
    size_t off_wc4 = off; off += al((size_t)24 * 48 * 256 * 4);
    // aliases (lifetimes disjoint): A2 over S, A3/A4 over abuf
    size_t off_A2 = off_S;
    size_t off_A3 = off_a;
    size_t off_A4 = off_a + (size_t)32 * 1024 * 1024;

    float*          ybuf = (float*)(ws + off_y);
    float*          abuf = (float*)(ws + off_a);
    unsigned short* Sbuf = (unsigned short*)(ws + off_S);
    float*          b1o  = (float*)(ws + off_b1);
    float*          b2o  = (float*)(ws + off_b2);
    float*          b3o  = (float*)(ws + off_b3);
    unsigned short* A2   = (unsigned short*)(ws + off_A2);
    unsigned short* A3   = (unsigned short*)(ws + off_A3);
    unsigned short* A4   = (unsigned short*)(ws + off_A4);

    dim3 blk(256);
    // ---- fc conv ----
    {
        int TOT = 16 * 48 * 224 * 224;
        k_conv_fc<<<(TOT + 255) / 256, blk, 0, stream>>>(x, fcw, fcb, ybuf);
    }
    // ---- region branches ----
    auto run_branch = [&](const float* in, int base, int C, int O, int gw, int R,
                          int tile, int Kpad, int Opad, unsigned int* Wf, float* bout) {
        int TOT = 16 * C * 224 * 224;
        k_branch_pre<<<(TOT + 255) / 256, blk, 0, stream>>>(
            in, br(base, 2), br(base, 0), br(base, 1), br(base, 3),
            abuf, Sbuf, C, gw, tile, Kpad);
        k_branch_dw<<<(TOT + 255) / 256, blk, 0, stream>>>(
            abuf, br(base, 8), br(base, 6), br(base, 4), br(base, 5), br(base, 7),
            Sbuf, C, gw, tile, Kpad);
        int KT = Kpad / 32, NT = Opad / 16;
        int ptot = R * KT * NT * 256;
        k_prepack_branch<<<(ptot + 255) / 256, blk, 0, stream>>>(
            br(base, 10), br(base, 12), Wf, R, C, O, Kpad, Opad);
        int waves = 16 * R * (tile * tile / 16) * NT;
        k_gemm_branch<<<(waves + 7) / 8, blk, 0, stream>>>(
            Sbuf, Wf, br(base, 9), br(base, 11), bout, R, C, O, Kpad, Opad, gw, tile);
    };
    run_branch(ybuf, B1, 48, 24, 8, 64, 28,  96, 32, (unsigned int*)(ws + off_wf1), b1o);
    run_branch(b1o,  B2, 24, 12, 4, 16, 56,  64, 16, (unsigned int*)(ws + off_wf2), b2o);
    run_branch(b2o,  B3, 12, 12, 2,  4, 112, 32, 16, (unsigned int*)(ws + off_wf3), b3o);

    // ---- combine + im2col-free A2 ----
    {
        int TOT = 16 * 48 * 224 * 224;
        k_combine<<<(TOT + 255) / 256, blk, 0, stream>>>(
            ybuf, b1o, b2o, b3o, n1s, n1b, n1m, n1v, A2);
    }
    // ---- prepack patch-conv weights ----
    k_prepack_conv<<<(24 * 12 * 256 + 255) / 256, blk, 0, stream>>>(
        c2w, (unsigned int*)(ws + off_wc2), 48, 4, 4, 192);
    k_prepack_conv<<<(24 * 12 * 256 + 255) / 256, blk, 0, stream>>>(
        c3w, (unsigned int*)(ws + off_wc3), 192, 2, 2, 192);
    k_prepack_conv<<<(24 * 48 * 256 + 255) / 256, blk, 0, stream>>>(
        c4w, (unsigned int*)(ws + off_wc4), 192, 2, 2, 768);
    // ---- tail GEMMs (LDS-staged, 4 N-tiles/wave) ----
    {
        // conv2: M=3136 (MT=196, MTG=25), N=192 (NTG=3)
        k_gemm_tail2<<<16 * 3 * 25, blk, 0, stream>>>(
            A2, (unsigned int*)(ws + off_wc2), bn2s, bn2b, bn2m, bn2v,
            A3, nullptr, 3136, 196, 192, 0, 56, 28, 784, 3136, 25);
    }
    {
        // conv3: M=784 (MT=49, MTG=7), N=192 (NTG=3)
        k_gemm_tail2<<<16 * 3 * 7, blk, 0, stream>>>(
            A3, (unsigned int*)(ws + off_wc3), bn3s, bn3b, bn3m, bn3v,
            A4, nullptr, 784, 49, 192, 0, 28, 14, 208, 784, 7);
    }
    {
        // conv4: M=208 (MT=13, MTG=2), N=768 (NTG=12)
        k_gemm_tail2<<<16 * 12 * 2, blk, 0, stream>>>(
            A4, (unsigned int*)(ws + off_wc4), bn4s, bn4b, bn4m, bn4v,
            nullptr, (float*)d_out, 208, 13, 768, 1, 14, 0, 0, 196, 2);
    }
}
